// NaiveAttention_54013508714833
// MI455X (gfx1250) — matrix operand order repaired and checked by reading, not compiled, not run on hardware
//
#include <hip/hip_runtime.h>

#define B_  2
#define L_  2048
#define D_  2048
#define H_  16
#define HD_ 128

typedef _Float16 f16;
typedef __attribute__((ext_vector_type(16))) _Float16 v16h;
typedef __attribute__((ext_vector_type(8)))  float    v8f;

union Frag {
  v16h h;
  uint4 q[2];
};
union H4 {
  f16 h[4];
  uint2 u;
};

__device__ __forceinline__ float red_max16(float v) {
  v = fmaxf(v, __shfl_xor(v, 1));
  v = fmaxf(v, __shfl_xor(v, 2));
  v = fmaxf(v, __shfl_xor(v, 4));
  v = fmaxf(v, __shfl_xor(v, 8));
  return v;
}
__device__ __forceinline__ float red_sum16(float v) {
  v += __shfl_xor(v, 1);
  v += __shfl_xor(v, 2);
  v += __shfl_xor(v, 4);
  v += __shfl_xor(v, 8);
  return v;
}

// gfx1250 async copy: 16 bytes global -> LDS, tracked by ASYNCcnt.
// LDS generic addresses keep the byte offset in the low 32 bits, so the
// truncated pointer is the VDST LDS address.
__device__ __forceinline__ void async_copy_b128(void* lds_ptr, const void* gptr) {
  unsigned ldsoff = (unsigned)(unsigned long long)lds_ptr;
  asm volatile("global_load_async_to_lds_b128 %0, %1, off"
               :: "v"(ldsoff), "v"((unsigned long long)gptr)
               : "memory");
}
__device__ __forceinline__ void wait_async0() {
  asm volatile("s_wait_asynccnt 0" ::: "memory");
}

// ---------------------------------------------------------------------------
// Kernel 1 (streaming, float4-vectorized):
//   q16  = f16(q * 1/sqrt(hd))          [B,L,D]
//   k16  = f16(k)                       [B,L,D]
//   v16t = f16(v) transposed            [B,H,HD,L]  (so V^T tiles are contiguous)
//   wo16 = f16(wo)                      [D,D]
//   kh/vh outputs (f32, [B,H,L,HD])     written to d_out
// ---------------------------------------------------------------------------
__global__ __launch_bounds__(256)
void prep_kernel(const float* __restrict__ q, const float* __restrict__ k,
                 const float* __restrict__ v, const float* __restrict__ wo,
                 f16* __restrict__ q16, f16* __restrict__ k16,
                 f16* __restrict__ v16t, f16* __restrict__ wo16,
                 float* __restrict__ kh_out, float* __restrict__ vh_out) {
  const float scale = 0.08838834764831845f; // 1/sqrt(128)
  long e4 = (long)blockIdx.x * blockDim.x + threadIdx.x;
  const long N4 = (long)B_ * L_ * D_ / 4;
  if (e4 >= N4) return;
  long e = e4 * 4;

  float4 qv = *(const float4*)(q + e);
  float4 kv = *(const float4*)(k + e);
  float4 vv = *(const float4*)(v + e);

  H4 qh, kh16;
  qh.h[0] = (f16)(qv.x * scale); qh.h[1] = (f16)(qv.y * scale);
  qh.h[2] = (f16)(qv.z * scale); qh.h[3] = (f16)(qv.w * scale);
  kh16.h[0] = (f16)kv.x; kh16.h[1] = (f16)kv.y;
  kh16.h[2] = (f16)kv.z; kh16.h[3] = (f16)kv.w;
  *(uint2*)(q16 + e) = qh.u;
  *(uint2*)(k16 + e) = kh16.u;

  int  c  = (int)(e % D_);
  long rb = e / D_;
  int  r  = (int)(rb % L_);
  int  b  = (int)(rb / L_);
  int  h  = c / HD_;
  int  d  = c % HD_;   // multiple of 4, same head for all 4 elements

  long o = (((long)(b * H_ + h) * L_ + r) * HD_ + d);
  *(float4*)(kh_out + o) = kv;
  *(float4*)(vh_out + o) = vv;

  // transposed f16 V: [(b,h), dim, seq]
  long vt = ((long)(b * H_ + h) * HD_ + d) * L_ + r;
  v16t[vt         ] = (f16)vv.x;
  v16t[vt + 1L * L_] = (f16)vv.y;
  v16t[vt + 2L * L_] = (f16)vv.z;
  v16t[vt + 3L * L_] = (f16)vv.w;

  if (e < (long)D_ * D_) {
    float4 wv = *(const float4*)(wo + e);
    H4 wh;
    wh.h[0] = (f16)wv.x; wh.h[1] = (f16)wv.y;
    wh.h[2] = (f16)wv.z; wh.h[3] = (f16)wv.w;
    *(uint2*)(wo16 + e) = wh.u;
  }
}

// ---------------------------------------------------------------------------
// Kernel 2: flash attention, 4 waves x 16 q rows, double-buffered async tiles.
// One __syncthreads per KV iteration; P round-trip is per-wave LDS (in-order).
// ---------------------------------------------------------------------------
__global__ __launch_bounds__(128)
void attn_kernel(const f16* __restrict__ q16, const f16* __restrict__ k16,
                 const f16* __restrict__ v16t, f16* __restrict__ attn16) {
  __shared__ f16 kT[2][64][128];    // [buf][kv_row][dim]   32 KB
  __shared__ f16 vT[2][128][64];    // [buf][dim][kv_row]   32 KB
  __shared__ f16 pT[4][16][64];     // per-wave P tile       8 KB

  const int tid  = threadIdx.x;
  const int wave = tid >> 5;
  const int lane = tid & 31;
  const int ln   = lane & 15;
  const int hi   = lane >> 4;

  const int mblk = blockIdx.x & 31;     // L_/64 q blocks
  const int bh   = blockIdx.x >> 5;
  const int h    = bh & (H_ - 1);
  const int b    = bh >> 4;

  const int m0 = mblk * 64;
  const int mw = m0 + wave * 16;

  const f16* kbase0 = k16  + (long)(b * L_) * D_ + h * HD_;
  const f16* vbase0 = v16t + (long)(b * H_ + h) * HD_ * L_;

  // issue async copies for KV block jb into buffer `buf`
  auto issue_tiles = [&](int jb, int buf) {
    const int j0 = jb * 64;
    const f16* kbase = kbase0 + (long)j0 * D_;   // 64 rows x 128 halves
#pragma unroll
    for (int i = 0; i < 8; ++i) {
      int idx = tid + i * 128;                   // 1024 16B chunks
      int rr = idx >> 4, cc = (idx & 15) * 8;
      async_copy_b128(&kT[buf][rr][cc], kbase + (long)rr * D_ + cc);
    }
    const f16* vbase = vbase0 + j0;              // 128 dims x 64 halves
#pragma unroll
    for (int i = 0; i < 8; ++i) {
      int idx = tid + i * 128;                   // 1024 16B chunks
      int dd = idx >> 3, cc = (idx & 7) * 8;
      async_copy_b128(&vT[buf][dd][cc], vbase + (long)dd * L_ + cc);
    }
  };

  // Q fragments: 16 rows x 128 dims as four 16x32 A-fragments (pre-scaled)
  Frag qa[4];
  {
    const f16* qrow = q16 + (long)(b * L_ + mw + ln) * D_ + h * HD_;
#pragma unroll
    for (int kc = 0; kc < 4; ++kc) {
      int k0 = kc * 32 + hi * 8;
      qa[kc].q[0] = *(const uint4*)(qrow + k0);
      qa[kc].q[1] = *(const uint4*)(qrow + k0 + 16);
    }
  }

  float m_i[8], l_i[8];
  v8f O[8];
#pragma unroll
  for (int r = 0; r < 8; ++r) { m_i[r] = -1e30f; l_i[r] = 0.f; }
#pragma unroll
  for (int d = 0; d < 8; ++d)
#pragma unroll
    for (int r = 0; r < 8; ++r) O[d][r] = 0.f;

  issue_tiles(0, 0);

  for (int jb = 0; jb <= mblk; ++jb) {
    const int j0  = jb * 64;
    const int buf = jb & 1;
    wait_async0();        // my async copies into `buf` complete
    __syncthreads();      // everyone's copies done; prior reads of `buf` done

    if (jb < mblk) issue_tiles(jb + 1, buf ^ 1);  // prefetch next block

    // S (16x64) = Q * K^T (scale pre-folded into Q)
    v8f S[4];
#pragma unroll
    for (int ns = 0; ns < 4; ++ns) {
      v8f acc;
#pragma unroll
      for (int r = 0; r < 8; ++r) acc[r] = 0.f;
#pragma unroll
      for (int kc = 0; kc < 4; ++kc) {
        Frag kb;
        const f16* krow = &kT[buf][ns * 16 + ln][0];
        int k0 = kc * 32 + hi * 8;
        kb.q[0] = *(const uint4*)(krow + k0);
        kb.q[1] = *(const uint4*)(krow + k0 + 16);
        acc = __builtin_amdgcn_wmma_f32_16x16x32_f16(
            false, qa[kc].h, false, kb.h, (short)0, acc, false, false);
      }
      S[ns] = acc;
    }

    // causal mask: only the diagonal block can have N > M
    if (jb == mblk) {
#pragma unroll
      for (int ns = 0; ns < 4; ++ns)
#pragma unroll
        for (int r = 0; r < 8; ++r) {
          int M = mw + r + 8 * hi;
          int N = j0 + ns * 16 + ln;
          if (N > M) S[ns][r] += -10000.f;
        }
    }

    // online softmax stats per row (row M = r + 8*hi, N across 16 lanes)
    float alpha[8];
#pragma unroll
    for (int r = 0; r < 8; ++r) {
      float mx = fmaxf(fmaxf(S[0][r], S[1][r]), fmaxf(S[2][r], S[3][r]));
      mx = red_max16(mx);
      float mnew = fmaxf(m_i[r], mx);
      alpha[r] = __expf(m_i[r] - mnew);
      float rs = 0.f;
#pragma unroll
      for (int ns = 0; ns < 4; ++ns) {
        float p = __expf(S[ns][r] - mnew);
        S[ns][r] = p;
        rs += p;
      }
      rs = red_sum16(rs);
      l_i[r] = l_i[r] * alpha[r] + rs;
      m_i[r] = mnew;
    }

    // rescale output accumulator
#pragma unroll
    for (int d = 0; d < 8; ++d)
#pragma unroll
      for (int r = 0; r < 8; ++r) O[d][r] *= alpha[r];

    // P: C-layout -> per-wave LDS -> A-layout (LDS ops in-order within wave)
#pragma unroll
    for (int ns = 0; ns < 4; ++ns)
#pragma unroll
      for (int r = 0; r < 8; ++r)
        pT[wave][r + 8 * hi][ns * 16 + ln] = (f16)S[ns][r];

    Frag pa[2];
#pragma unroll
    for (int kc = 0; kc < 2; ++kc) {
      const f16* prow = &pT[wave][ln][0];
      int k0 = kc * 32 + hi * 8;
      pa[kc].q[0] = *(const uint4*)(prow + k0);
      pa[kc].q[1] = *(const uint4*)(prow + k0 + 16);
    }

    // O (16x128) += P (16x64) * V (64x128)
#pragma unroll
    for (int d = 0; d < 8; ++d) {
#pragma unroll
      for (int kc = 0; kc < 2; ++kc) {
        Frag vb;
        const f16* vrow = &vT[buf][d * 16 + ln][0];
        int k0 = kc * 32 + hi * 8;
        vb.q[0] = *(const uint4*)(vrow + k0);
        vb.q[1] = *(const uint4*)(vrow + k0 + 16);
        O[d] = __builtin_amdgcn_wmma_f32_16x16x32_f16(
            false, pa[kc].h, false, vb.h, (short)0, O[d], false, false);
      }
    }
  }

  // normalize and store attention output (f16, [B, L, D] layout)
#pragma unroll
  for (int d = 0; d < 8; ++d) {
#pragma unroll
    for (int r = 0; r < 8; ++r) {
      int M   = mw + r + 8 * hi;
      int col = h * HD_ + d * 16 + ln;
      attn16[(long)(b * L_ + M) * D_ + col] = (f16)(O[d][r] / l_i[r]);
    }
  }
}

// ---------------------------------------------------------------------------
// Kernel 3: out[i][n] = sum_k attn[i][k] * wo[n][k]   (x @ Wo^T)
// 128x128 block tile, 8 waves (32x64 each), double-buffered async K tiles.
// ---------------------------------------------------------------------------
__global__ __launch_bounds__(256)
void proj_kernel(const f16* __restrict__ x, const f16* __restrict__ w,
                 float* __restrict__ out) {
  __shared__ f16 aT[2][128][64];   // 32 KB
  __shared__ f16 bT[2][128][64];   // 32 KB (wo rows directly: [n][k])

  const int tid  = threadIdx.x;
  const int wave = tid >> 5;
  const int lane = tid & 31;
  const int ln   = lane & 15;
  const int hi   = lane >> 4;

  const int nb = blockIdx.x & 15;   // D/128
  const int mb = blockIdx.x >> 4;   // (B*L)/128
  const int i0 = mb * 128, n0 = nb * 128;
  const int wm = (wave & 3) * 32;
  const int wn = (wave >> 2) * 64;

  auto issue_tiles = [&](int kb, int buf) {
    const int k0 = kb * 64;
#pragma unroll
    for (int i = 0; i < 4; ++i) {
      int idx = tid + i * 256;                 // 1024 16B chunks per tile
      int rr = idx >> 3, cc = (idx & 7) * 8;
      async_copy_b128(&aT[buf][rr][cc], x + (long)(i0 + rr) * D_ + k0 + cc);
      async_copy_b128(&bT[buf][rr][cc], w + (long)(n0 + rr) * D_ + k0 + cc);
    }
  };

  v8f acc[2][4];
#pragma unroll
  for (int ms = 0; ms < 2; ++ms)
#pragma unroll
    for (int ns = 0; ns < 4; ++ns)
#pragma unroll
      for (int r = 0; r < 8; ++r) acc[ms][ns][r] = 0.f;

  issue_tiles(0, 0);

  const int KB = D_ / 64;
  for (int kb = 0; kb < KB; ++kb) {
    const int buf = kb & 1;
    wait_async0();
    __syncthreads();
    if (kb + 1 < KB) issue_tiles(kb + 1, buf ^ 1);

#pragma unroll
    for (int kc = 0; kc < 2; ++kc) {
      Frag af[2];
#pragma unroll
      for (int ms = 0; ms < 2; ++ms) {
        const f16* arow = &aT[buf][wm + ms * 16 + ln][0];
        int kk = kc * 32 + hi * 8;
        af[ms].q[0] = *(const uint4*)(arow + kk);
        af[ms].q[1] = *(const uint4*)(arow + kk + 16);
      }
#pragma unroll
      for (int ns = 0; ns < 4; ++ns) {
        Frag bf;
        const f16* brow = &bT[buf][wn + ns * 16 + ln][0];
        int kk = kc * 32 + hi * 8;
        bf.q[0] = *(const uint4*)(brow + kk);
        bf.q[1] = *(const uint4*)(brow + kk + 16);
#pragma unroll
        for (int ms = 0; ms < 2; ++ms)
          acc[ms][ns] = __builtin_amdgcn_wmma_f32_16x16x32_f16(
              false, af[ms].h, false, bf.h, (short)0, acc[ms][ns], false, false);
      }
    }
  }

#pragma unroll
  for (int ms = 0; ms < 2; ++ms)
#pragma unroll
    for (int ns = 0; ns < 4; ++ns)
#pragma unroll
      for (int r = 0; r < 8; ++r) {
        int M = i0 + wm + ms * 16 + r + 8 * hi;
        int N = n0 + wn + ns * 16 + ln;
        out[(long)M * D_ + N] = acc[ms][ns][r];
      }
}

// ---------------------------------------------------------------------------
extern "C" void kernel_launch(void* const* d_in, const int* in_sizes, int n_in,
                              void* d_out, int out_size, void* d_ws, size_t ws_size,
                              hipStream_t stream) {
  const float* q  = (const float*)d_in[0];
  const float* k  = (const float*)d_in[1];
  const float* v  = (const float*)d_in[2];
  const float* wo = (const float*)d_in[3];

  float* out = (float*)d_out;
  float* kh  = out + (size_t)B_ * L_ * D_;
  float* vh  = kh + (size_t)B_ * H_ * L_ * HD_;

  char*  ws  = (char*)d_ws;
  const size_t sz = (size_t)B_ * L_ * D_ * sizeof(f16);   // 16 MB per tensor
  f16* q16    = (f16*)(ws);
  f16* k16    = (f16*)(ws + sz);
  f16* v16t   = (f16*)(ws + 2 * sz);
  f16* wo16   = (f16*)(ws + 3 * sz);
  f16* attn16 = (f16*)(ws + 3 * sz + (size_t)D_ * D_ * sizeof(f16));

  const long n4 = (long)B_ * L_ * D_ / 4;
  prep_kernel<<<(int)((n4 + 255) / 256), 256, 0, stream>>>(
      q, k, v, wo, q16, k16, v16t, wo16, kh, vh);
  attn_kernel<<<B_ * H_ * (L_ / 64), 128, 0, stream>>>(q16, k16, v16t, attn16);
  proj_kernel<<<(B_ * L_ / 128) * (D_ / 128), 256, 0, stream>>>(attn16, wo16, out);
}
